// RWKV7Attention_50311246905930
// MI455X (gfx1250) — matrix operand order repaired and checked by reading, not compiled, not run on hardware
//
#include <hip/hip_runtime.h>
#include <hip/hip_bf16.h>
#include <math.h>

// ---------------------------------------------------------------------------
// RWKV7 attention block for MI455X (gfx1250, wave32, WMMA).
// GEMMs: bf16 v_wmma_f32_16x16x32_bf16, LDS double-buffered K pipeline,
// async global->LDS staging (ASYNCcnt) where layout is contiguous, token-shift
// mixing fused into the A-operand stage for all QKV/LoRA-down projections.
// ---------------------------------------------------------------------------

typedef __bf16 bf16_t;
typedef __attribute__((ext_vector_type(16))) __bf16 v16bf;
typedef __attribute__((ext_vector_type(8)))  __bf16 v8bf;
typedef __attribute__((ext_vector_type(8)))  float  v8f;

#define BM 128
#define BN 128
#define BK 32
#define A_STRIDE 40    // 32 + 8 bf16 pad (80B rows, 16B aligned)
#define B_STRIDE 136   // 128 + 8 bf16 pad (272B rows, 16B aligned)

static __device__ __forceinline__ v16bf cat8(v8bf lo, v8bf hi) {
  v16bf r;
#pragma unroll
  for (int i = 0; i < 8; ++i) { r[i] = lo[i]; r[i + 8] = hi[i]; }
  return r;
}

// LDS byte offset of a __shared__ object (addrspace(3) ptrs are 32-bit).
static __device__ __forceinline__ unsigned lds_offset(const void* p) {
  return (unsigned)(unsigned long long)
      (const __attribute__((address_space(3))) char*)p;
}

// Async DMA: 16B global -> LDS, tracked by ASYNCcnt (CDNA5 ISA).
static __device__ __forceinline__ void async_g2l_b128(unsigned lds_byte_off,
                                                      const void* gaddr) {
  asm volatile("global_load_async_to_lds_b128 %0, %1, off"
               :: "v"(lds_byte_off),
                  "v"((unsigned long long)gaddr)
               : "memory");
}
static __device__ __forceinline__ void wait_asynccnt0() {
  asm volatile("s_wait_asynccnt 0" ::: "memory");
}

// ---------------------------------------------------------------------------
__global__ __launch_bounds__(256) void f32_to_bf16_kernel(
    const float* __restrict__ x, bf16_t* __restrict__ y, int n) {
  int i = blockIdx.x * 256 + threadIdx.x;
  if (i < n) y[i] = (bf16_t)x[i];
}

// ---------------------------------------------------------------------------
// C[M,N] = act(A[M,K] @ W[N,K]^T + bias)
// mix != nullptr: A generated on the fly from f32 hidden with token-shift
//   mixing a[m,k] = h[m,k] + (h[m-1,k] - h[m,k]) * mix[k]   (h[-1] = 0 per seq)
// mix == nullptr: A is a plain bf16 [M,K] matrix (async-staged when full tile).
// act: 0 = none, 1 = tanh, 2 = sigmoid.  Cb != nullptr -> bf16 out else f32.
__global__ __launch_bounds__(256) void gemm_nt_wmma(
    const void* __restrict__ Aptr, const float* __restrict__ mix, int T,
    const bf16_t* __restrict__ W, int M, int N, int K,
    float* __restrict__ Cf, bf16_t* __restrict__ Cb,
    const float* __restrict__ bias, int act)
{
  __shared__ __align__(16) bf16_t As[2][BM * A_STRIDE];
  __shared__ __align__(16) bf16_t Bs[2][BK * B_STRIDE];

  const int tid  = threadIdx.x;
  const int lane = tid & 31;
  const int wave = tid >> 5;
  const int wm   = wave >> 2;      // 2 wave-rows (M)
  const int wn   = wave & 3;       // 4 wave-cols (N)
  const int m0   = blockIdx.x * BM;
  const int n0   = blockIdx.y * BN;

  const int srow  = tid >> 1;             // staging row 0..127
  const int scol0 = (tid & 1) * 16;       // staging col half 0/16
  const bool fullM = (m0 + BM <= M);

  v8f acc[4][2] = {};

  const int mlane = lane & 15;
  const int khalf = lane >> 4;            // wave32 A-fragment half select

  // ---- tile staging (fast branch-free path when tile fully in range) ----
  auto stageA = [&](int buf, int k0) {
    const int gm = m0 + srow;
    const bool fullK = (k0 + BK <= K);
    if (mix) {
      const float* Hh = (const float*)Aptr;
      if (fullM & fullK) {
        const int  t       = gm % T;
        const bool hasPrev = (t > 0);
        const float*  cur  = Hh + (size_t)gm * K + k0 + scol0;
        const float*  prv  = cur - K;
        const float*  mxp  = mix + k0 + scol0;
        if (k0 + 2 * BK <= K) __builtin_prefetch(cur + 2 * BK, 0, 0);
        float vals[16];
#pragma unroll
        for (int q = 0; q < 4; ++q) {
          const float4 xc = ((const float4*)cur)[q];
          const float4 mx = ((const float4*)mxp)[q];
          float4 xp = make_float4(0.f, 0.f, 0.f, 0.f);
          if (hasPrev) xp = ((const float4*)prv)[q];
          vals[q * 4 + 0] = xc.x + (xp.x - xc.x) * mx.x;
          vals[q * 4 + 1] = xc.y + (xp.y - xc.y) * mx.y;
          vals[q * 4 + 2] = xc.z + (xp.z - xc.z) * mx.z;
          vals[q * 4 + 3] = xc.w + (xp.w - xc.w) * mx.w;
        }
        v8bf o0, o1;
#pragma unroll
        for (int i = 0; i < 8; ++i) {
          o0[i] = (bf16_t)vals[i];
          o1[i] = (bf16_t)vals[8 + i];
        }
        *(v8bf*)&As[buf][srow * A_STRIDE + scol0]     = o0;
        *(v8bf*)&As[buf][srow * A_STRIDE + scol0 + 8] = o1;
      } else {
        const int  t       = (gm < M) ? (gm % T) : 0;
        const bool hasPrev = (gm < M) && (t > 0);
#pragma unroll
        for (int i = 0; i < 16; ++i) {
          const int gk = k0 + scol0 + i;
          float v = 0.f;
          if (gm < M && gk < K) {
            const float mx = mix[gk];
            const float x  = Hh[(size_t)gm * K + gk];
            const float xp = hasPrev ? Hh[(size_t)(gm - 1) * K + gk] : 0.f;
            v = x + (xp - x) * mx;
          }
          As[buf][srow * A_STRIDE + scol0 + i] = (bf16_t)v;
        }
      }
    } else {
      const bf16_t* Ab = (const bf16_t*)Aptr;
      if (fullM & fullK) {
        // ASYNCcnt-tracked DMA straight into the LDS tile (32B/thread).
        const bf16_t* gsrc = Ab + (size_t)gm * K + k0 + scol0;
        const unsigned ldst =
            lds_offset(&As[buf][srow * A_STRIDE + scol0]);
        async_g2l_b128(ldst,      gsrc);
        async_g2l_b128(ldst + 16, gsrc + 8);
      } else {
#pragma unroll
        for (int i = 0; i < 16; ++i) {
          const int gk = k0 + scol0 + i;
          As[buf][srow * A_STRIDE + scol0 + i] =
              (gm < M && gk < K) ? Ab[(size_t)gm * K + gk] : (bf16_t)0.f;
        }
      }
    }
  };

  auto stageB = [&](int buf, int k0) {
    const int gn = n0 + srow;
    const bool fullK = (k0 + BK <= K);
    if ((gn < N) & fullK) {
      const bf16_t* src = W + (size_t)gn * K + k0 + scol0;
      if (k0 + 2 * BK <= K) __builtin_prefetch(src + 2 * BK, 0, 0);
      const v8bf b0 = *(const v8bf*)src;
      const v8bf b1 = *(const v8bf*)(src + 8);
#pragma unroll
      for (int i = 0; i < 8; ++i) {
        Bs[buf][(scol0 + i)     * B_STRIDE + srow] = b0[i];
        Bs[buf][(scol0 + 8 + i) * B_STRIDE + srow] = b1[i];
      }
    } else {
#pragma unroll
      for (int i = 0; i < 16; ++i) {
        const int gk = k0 + scol0 + i;
        Bs[buf][(scol0 + i) * B_STRIDE + srow] =
            (gn < N && gk < K) ? W[(size_t)gn * K + gk] : (bf16_t)0.f;
      }
    }
  };

  // ---- double-buffered K pipeline ----
  const int nK = (K + BK - 1) / BK;
  stageA(0, 0);
  stageB(0, 0);
  wait_asynccnt0();
  __syncthreads();

  for (int kb = 0; kb < nK; ++kb) {
    const int cur = kb & 1;
    if (kb + 1 < nK) {            // prefetch next tile into the other buffer
      stageA(cur ^ 1, (kb + 1) * BK);
      stageB(cur ^ 1, (kb + 1) * BK);
    }

    // A 16x32 bf16 frag: lane = M%16 | khalf; VGPR0-3 = K(8*khalf..+7), +16
    v16bf afrag[4], bfrag[2];
#pragma unroll
    for (int i = 0; i < 4; ++i) {
      const bf16_t* p =
          &As[cur][(wm * 64 + i * 16 + mlane) * A_STRIDE + khalf * 8];
      afrag[i] = cat8(*(const v8bf*)p, *(const v8bf*)(p + 16));
    }
    // B 32x16 bf16 frag: lane = N%16 | khalf; elems K(8*khalf..+7), +16
#pragma unroll
    for (int j = 0; j < 2; ++j) {
      const bf16_t* p = &Bs[cur][(khalf * 8) * B_STRIDE + wn * 32 + j * 16 + mlane];
#pragma unroll
      for (int s = 0; s < 8; ++s) {
        bfrag[j][s]     = p[s * B_STRIDE];
        bfrag[j][s + 8] = p[(s + 16) * B_STRIDE];
      }
    }

#pragma unroll
    for (int i = 0; i < 4; ++i)
#pragma unroll
      for (int j = 0; j < 2; ++j)
        acc[i][j] = __builtin_amdgcn_wmma_f32_16x16x32_bf16(
            false, afrag[i], false, bfrag[j], (short)0, acc[i][j], false, false);

    wait_asynccnt0();             // next tile's DMA landed
    __syncthreads();
  }

  // ---- epilogue: C 16x16 f32 layout: lane = N%16 | Mhigh, VGPR r = M row ----
  const int nlo = lane & 15;
  const int mhi = (lane >> 4) * 8;
#pragma unroll
  for (int i = 0; i < 4; ++i)
#pragma unroll
    for (int j = 0; j < 2; ++j)
#pragma unroll
      for (int rr = 0; rr < 8; ++rr) {
        const int gm = m0 + wm * 64 + i * 16 + mhi + rr;
        const int gn = n0 + wn * 32 + j * 16 + nlo;
        if (gm < M && gn < N) {
          float v = acc[i][j][rr];
          if (bias) v += bias[gn];
          if (act == 1)      v = tanhf(v);
          else if (act == 2) v = 1.f / (1.f + __expf(-v));
          if (Cb) Cb[(size_t)gm * N + gn] = (bf16_t)v;
          else    Cf[(size_t)gm * N + gn] = v;
        }
      }
}

// ---------------------------------------------------------------------------
// Fused pre-scan elementwise pass (one block per token row, 8 ch/thread).
//   kbuf: in xk@Wk^T          -> out k*(1+(a-1)*k_a)
//   vbuf: in xv@Wv^T          -> out lerp(v, v_state, sigmoid(v_lora))
//   wbuf: in w-lora (pre-act) -> out exp(-softplus(-wl)-0.5) = sig(wl)*e^-0.5
//   nbuf: in v-lora           -> out -kk_hat        (scan "a" operand)
//   abuf: in a-lora           -> out kk_hat*sig(al) (scan "b" operand)
__global__ __launch_bounds__(256) void fuse_pre_kernel(
    const float* __restrict__ v_state, const float* __restrict__ k_k,
    const float* __restrict__ k_a,
    float* __restrict__ kbuf, float* __restrict__ vbuf,
    float* __restrict__ wbuf, float* __restrict__ nbuf,
    float* __restrict__ abuf, int D)
{
  __shared__ float s1[256];
  const int m = blockIdx.x, tid = threadIdx.x;
  const int c0 = tid * 8;
  const size_t base = (size_t)m * D + c0;

  float kk[8];
  float ss = 0.f;
#pragma unroll
  for (int i = 0; i < 8; ++i) {
    const float t = kbuf[base + i] * k_k[c0 + i];
    kk[i] = t; ss += t * t;
  }
  s1[tid] = ss;
  __syncthreads();
  const int hb = (tid >> 3) << 3;   // 8 threads per 64-ch head
  float tot = 0.f;
#pragma unroll
  for (int u = 0; u < 8; ++u) tot += s1[hb + u];
  const float inv = 1.f / fmaxf(sqrtf(tot), 1e-12f);

#pragma unroll
  for (int i = 0; i < 8; ++i) {
    const int c = c0 + i;
    const float a  = 1.f / (1.f + __expf(-abuf[base + i]));
    const float sv = 1.f / (1.f + __expf(-nbuf[base + i]));
    const float wd = (1.f / (1.f + __expf(-wbuf[base + i]))) * 0.60653065971f;
    const float kt = kbuf[base + i];
    const float kh = kk[i] * inv;
    kbuf[base + i] = kt * (1.f + (a - 1.f) * k_a[c]);
    vbuf[base + i] = vbuf[base + i] + sv * (v_state[base + i] - vbuf[base + i]);
    wbuf[base + i] = wd;
    nbuf[base + i] = -kh;
    abuf[base + i] = kh * a;
  }
}

// ---------------------------------------------------------------------------
// Sequential DPLR scan: one block per (b,h); lane j owns state column S[:,j].
__global__ __launch_bounds__(64) void rwkv_scan_kernel(
    const float* __restrict__ r, const float* __restrict__ wd,
    const float* __restrict__ k, const float* __restrict__ v,
    const float* __restrict__ na, const float* __restrict__ bb,
    float* __restrict__ o, int T, int H, int D)
{
  const int b = blockIdx.x / H, h = blockIdx.x % H;
  const int j = threadIdx.x;
  __shared__ float lr[64], lw[64], lk[64], la[64], lb[64];
  float S[64];
#pragma unroll
  for (int i = 0; i < 64; ++i) S[i] = 0.f;

  size_t idx = ((size_t)b * T) * D + (size_t)h * 64 + j;
  for (int t = 0; t < T; ++t, idx += D) {
    lr[j] = r[idx]; lw[j] = wd[idx]; lk[j] = k[idx];
    la[j] = na[idx]; lb[j] = bb[idx];
    const float vj = v[idx];
    __syncthreads();
    float sa = 0.f;
#pragma unroll
    for (int q = 0; q < 64; ++q) { const float s = S[q] * lw[q]; S[q] = s; sa += la[q] * s; }
    float oj = 0.f;
#pragma unroll
    for (int q = 0; q < 64; ++q) {
      const float s = S[q] + lb[q] * sa + lk[q] * vj;
      S[q] = s; oj += lr[q] * s;
    }
    o[idx] = oj;
    __syncthreads();
  }
}

// ---------------------------------------------------------------------------
// Per-head GroupNorm + (r*k*r_k).sum * v term + gate; emits bf16 z for Wo GEMM.
__global__ __launch_bounds__(256) void gn_gate_kernel(
    const float* __restrict__ o, const float* __restrict__ r,
    const float* __restrict__ k, const float* __restrict__ v,
    const float* __restrict__ g, const float* __restrict__ r_k,
    const float* __restrict__ gn_w, const float* __restrict__ gn_b,
    bf16_t* __restrict__ z, int D, float eps)
{
  __shared__ float s1[256], s2[256], s3[256];
  const int m = blockIdx.x, tid = threadIdx.x;
  const int c0 = tid * 8;
  const size_t base = (size_t)m * D + c0;

  float oc[8];
  float ps = 0.f, ps2 = 0.f, pd = 0.f;
#pragma unroll
  for (int i = 0; i < 8; ++i) {
    const float x = o[base + i];
    oc[i] = x; ps += x; ps2 += x * x;
    pd += r[base + i] * k[base + i] * r_k[c0 + i];
  }
  s1[tid] = ps; s2[tid] = ps2; s3[tid] = pd;
  __syncthreads();
  const int hb = (tid >> 3) << 3;
  float sm = 0.f, sq = 0.f, dt = 0.f;
#pragma unroll
  for (int u = 0; u < 8; ++u) { sm += s1[hb + u]; sq += s2[hb + u]; dt += s3[hb + u]; }
  const float mu   = sm * (1.f / 64.f);
  const float var  = sq * (1.f / 64.f) - mu * mu;
  const float rstd = rsqrtf(var + eps);

#pragma unroll
  for (int i = 0; i < 8; ++i) {
    const int c = c0 + i;
    float on = (oc[i] - mu) * rstd * gn_w[c] + gn_b[c];
    on += dt * v[base + i];
    z[base + i] = (bf16_t)(on * g[base + i]);
  }
}

// ---------------------------------------------------------------------------
extern "C" void kernel_launch(void* const* d_in, const int* in_sizes, int n_in,
                              void* d_out, int out_size, void* d_ws, size_t ws_size,
                              hipStream_t stream)
{
  const float* hidden  = (const float*)d_in[0];
  const float* v_state = (const float*)d_in[1];
  const float* x_x     = (const float*)d_in[2];
  const float* k_k     = (const float*)d_in[3];
  const float* k_a     = (const float*)d_in[4];
  const float* r_k     = (const float*)d_in[5];
  const float* Wr      = (const float*)d_in[6];
  const float* Wk      = (const float*)d_in[7];
  const float* Wv      = (const float*)d_in[8];
  const float* Wo      = (const float*)d_in[9];
  const float* wA      = (const float*)d_in[10];
  const float* wB      = (const float*)d_in[11];
  const float* wb      = (const float*)d_in[12];
  const float* vA      = (const float*)d_in[13];
  const float* vB      = (const float*)d_in[14];
  const float* vb      = (const float*)d_in[15];
  const float* aA      = (const float*)d_in[16];
  const float* aB      = (const float*)d_in[17];
  const float* ab      = (const float*)d_in[18];
  const float* gA      = (const float*)d_in[19];
  const float* gB      = (const float*)d_in[20];
  const float* gn_w    = (const float*)d_in[21];
  const float* gn_b    = (const float*)d_in[22];

  const int D  = in_sizes[3];              // 2048
  const int M  = in_sizes[0] / D;          // B*T = 4096
  const int T  = (M % 2048 == 0) ? 2048 : M;
  const int Bv = M / T;
  const int H  = D / 64;
  const int RW = in_sizes[10] / D;         // 64
  const int RV = in_sizes[13] / D;         // 16
  const int RA = in_sizes[16] / D;         // 64
  const int RG = in_sizes[19] / D;         // 128
  const float eps = 64.0f * 1e-5f;         // GN_EPS = DH * 1e-5

  // ---- workspace carve-out (~320MB) ----
  char* ws = (char*)d_ws;
  size_t off = 0;
  auto take = [&](size_t bytes) -> char* {
    char* p = ws + off;
    off = (off + bytes + 255) & ~(size_t)255;
    return p;
  };
  const size_t MD = (size_t)M * D;
  float* rbuf  = (float*)take(MD * 4);
  float* kbuf  = (float*)take(MD * 4);
  float* vbuf  = (float*)take(MD * 4);
  float* wlbuf = (float*)take(MD * 4);
  float* vlbuf = (float*)take(MD * 4);
  float* albuf = (float*)take(MD * 4);
  float* gbuf  = (float*)take(MD * 4);
  float* obuf  = (float*)take(MD * 4);
  bf16_t* zbuf = (bf16_t*)take(MD * 2);
  bf16_t* bWr  = (bf16_t*)take((size_t)D * D * 2);
  bf16_t* bWk  = (bf16_t*)take((size_t)D * D * 2);
  bf16_t* bWv  = (bf16_t*)take((size_t)D * D * 2);
  bf16_t* bWo  = (bf16_t*)take((size_t)D * D * 2);
  bf16_t* bwA  = (bf16_t*)take((size_t)RW * D * 2);
  bf16_t* bwB  = (bf16_t*)take((size_t)D * RW * 2);
  bf16_t* bvA  = (bf16_t*)take((size_t)RV * D * 2);
  bf16_t* bvB  = (bf16_t*)take((size_t)D * RV * 2);
  bf16_t* baA  = (bf16_t*)take((size_t)RA * D * 2);
  bf16_t* baB  = (bf16_t*)take((size_t)D * RA * 2);
  bf16_t* bgA  = (bf16_t*)take((size_t)RG * D * 2);
  bf16_t* bgB  = (bf16_t*)take((size_t)D * RG * 2);
  bf16_t* hw   = (bf16_t*)take((size_t)M * RW * 2);
  bf16_t* hv   = (bf16_t*)take((size_t)M * RV * 2);
  bf16_t* ha   = (bf16_t*)take((size_t)M * RA * 2);
  bf16_t* hg   = (bf16_t*)take((size_t)M * RG * 2);
  (void)ws_size; (void)n_in; (void)out_size;

  auto conv = [&](const float* src, bf16_t* dst, size_t n) {
    f32_to_bf16_kernel<<<dim3((unsigned)((n + 255) / 256)), 256, 0, stream>>>(
        src, dst, (int)n);
  };
  conv(Wr, bWr, (size_t)D * D); conv(Wk, bWk, (size_t)D * D);
  conv(Wv, bWv, (size_t)D * D); conv(Wo, bWo, (size_t)D * D);
  conv(wA, bwA, (size_t)RW * D); conv(wB, bwB, (size_t)D * RW);
  conv(vA, bvA, (size_t)RV * D); conv(vB, bvB, (size_t)D * RV);
  conv(aA, baA, (size_t)RA * D); conv(aB, baB, (size_t)D * RA);
  conv(gA, bgA, (size_t)RG * D); conv(gB, bgB, (size_t)D * RG);

  auto gemm = [&](const void* A, const float* mixrow, const bf16_t* Wb,
                  int m, int n, int kk, float* cf, bf16_t* cb,
                  const float* bias, int act) {
    dim3 grid((unsigned)((m + BM - 1) / BM), (unsigned)((n + BN - 1) / BN));
    gemm_nt_wmma<<<grid, 256, 0, stream>>>(A, mixrow, T, Wb, m, n, kk,
                                           cf, cb, bias, act);
  };

  // ---- stage 1: mixed-input GEMMs (token-shift mix fused into A loads) ----
  gemm(hidden, x_x + 0 * D, bWr, M, D,  D, rbuf,  nullptr, nullptr, 0); // r
  gemm(hidden, x_x + 2 * D, bWk, M, D,  D, kbuf,  nullptr, nullptr, 0); // k tmp
  gemm(hidden, x_x + 3 * D, bWv, M, D,  D, vbuf,  nullptr, nullptr, 0); // v tmp
  gemm(hidden, x_x + 1 * D, bwA, M, RW, D, nullptr, hw, nullptr, 1);    // tanh
  gemm(hidden, x_x + 3 * D, bvA, M, RV, D, nullptr, hv, nullptr, 0);
  gemm(hidden, x_x + 4 * D, baA, M, RA, D, nullptr, ha, nullptr, 0);
  gemm(hidden, x_x + 5 * D, bgA, M, RG, D, nullptr, hg, nullptr, 2);    // sigmoid

  // ---- stage 2: LoRA up-projections (async-staged bf16 A) ----
  gemm(hw, nullptr, bwB, M, D, RW, wlbuf, nullptr, wb, 0);
  gemm(hv, nullptr, bvB, M, D, RV, vlbuf, nullptr, vb, 0);
  gemm(ha, nullptr, baB, M, D, RA, albuf, nullptr, ab, 0);
  gemm(hg, nullptr, bgB, M, D, RG, gbuf,  nullptr, nullptr, 0);

  // ---- fused elementwise: decay/kk-norm/k/v/a/b operands ----
  fuse_pre_kernel<<<dim3((unsigned)M), 256, 0, stream>>>(
      v_state, k_k, k_a, kbuf, vbuf, wlbuf, vlbuf, albuf, D);

  // ---- sequential DPLR scan ----
  rwkv_scan_kernel<<<dim3((unsigned)(Bv * H)), 64, 0, stream>>>(
      rbuf, wlbuf, kbuf, vbuf, vlbuf, albuf, obuf, T, H, D);

  // ---- GroupNorm + bonus term + gate -> bf16 z ----
  gn_gate_kernel<<<dim3((unsigned)M), 256, 0, stream>>>(
      obuf, rbuf, kbuf, vbuf, gbuf, r_k, gn_w, gn_b, zbuf, D, eps);

  // ---- output projection (async-staged bf16 A) ----
  gemm(zbuf, nullptr, bWo, M, D, D, (float*)d_out, nullptr, nullptr, 0);
}
